// DistRegularizeLoss_8856222564952
// MI455X (gfx1250) — matrix operand order repaired and checked by reading, not compiled, not run on hardware
//
#include <hip/hip_runtime.h>
#include <hip/hip_bf16.h>

typedef float  v8f   __attribute__((ext_vector_type(8)));
typedef __bf16 v16bf __attribute__((ext_vector_type(16)));
typedef __bf16 v8bf  __attribute__((ext_vector_type(8)));
typedef __bf16 v4bf  __attribute__((ext_vector_type(4)));

#define DDIM 256
#define LDBF 264          // padded bf16 row stride: 264*2B = 132 dwords == 4 mod 64 banks
#define EPSF 1e-12f
#define WAVES 4           // waves per block; B tile shared across waves

// ---------------------------------------------------------------------------
// Zero the per-row accumulator and the output scalar (runs every call: the
// harness does not re-poison buffers between graph replays).
// ---------------------------------------------------------------------------
__global__ void init_kernel(float* __restrict__ d_accum, int n, float* __restrict__ out) {
    int i = blockIdx.x * blockDim.x + threadIdx.x;
    if (i < n) d_accum[i] = 0.0f;
    if (i == 0) out[0] = 0.0f;
}

// ---------------------------------------------------------------------------
// Row sums of squares: one 256-thread block per row of 256 contiguous floats.
// ---------------------------------------------------------------------------
__global__ void rownorm_kernel(const float* __restrict__ x, float* __restrict__ norms, int nrows) {
    __shared__ float red[256];
    int row = blockIdx.x;
    if (row >= nrows) return;
    float v = x[(size_t)row * DDIM + threadIdx.x];
    red[threadIdx.x] = v * v;
    __syncthreads();
    for (int s = 128; s > 0; s >>= 1) {
        if (threadIdx.x < s) red[threadIdx.x] += red[threadIdx.x + s];
        __syncthreads();
    }
    if (threadIdx.x == 0) norms[row] = red[0];
}

// ---------------------------------------------------------------------------
// Stage a 16x256 f32 tile into LDS as bf16 hi/lo split (x = hi + lo).
// Coalesced float4 global reads; packed v4bf (8B) LDS stores.
// ---------------------------------------------------------------------------
__device__ inline void stage_split(const float* __restrict__ src,  // row-major [16][256]
                                   __bf16* __restrict__ hi, __bf16* __restrict__ lo,
                                   int t, int nthreads) {
    for (int i = t; i < 16 * DDIM / 4; i += nthreads) {
        const int r  = i >> 6;              // 64 float4 per row
        const int c4 = i & 63;
        const float4 v = ((const float4*)(src + (size_t)r * DDIM))[c4];
        v4bf h, l;
        h[0] = (__bf16)v.x; l[0] = (__bf16)(v.x - (float)h[0]);
        h[1] = (__bf16)v.y; l[1] = (__bf16)(v.y - (float)h[1]);
        h[2] = (__bf16)v.z; l[2] = (__bf16)(v.z - (float)h[2]);
        h[3] = (__bf16)v.w; l[3] = (__bf16)(v.w - (float)h[3]);
        *(v4bf*)&hi[r * LDBF + c4 * 4] = h;   // byte addr 528r+8c4 -> 8B aligned
        *(v4bf*)&lo[r * LDBF + c4 * 4] = l;
    }
}

#define SHUF16(a, b) __builtin_shufflevector(a, b, 0,1,2,3,4,5,6,7,8,9,10,11,12,13,14,15)

// ---------------------------------------------------------------------------
// 4 waves per block; wave w owns n-tile (blockIdx.x*4+w)*16, all waves share
// each 16-row B tile. xy via bf16 hi/lo split: 3x v_wmma_f32_16x16x32_bf16
// per K=32 (A_hi*B_hi + A_hi*B_lo + A_lo*B_hi), f32 accumulate.
// ---------------------------------------------------------------------------
__global__ __launch_bounds__(32 * WAVES)
void dist_tile_kernel(const float* __restrict__ A,      // ide_feats [N,256]
                      const float* __restrict__ B,      // u_feats   [M,256]
                      const float* __restrict__ x2,
                      const float* __restrict__ y2,
                      float* __restrict__ d_accum,
                      int N, int M, int mtiles_per_chunk) {
    __shared__ __bf16 lAhi[WAVES][16 * LDBF];
    __shared__ __bf16 lAlo[WAVES][16 * LDBF];
    __shared__ __bf16 lBhi[16 * LDBF];
    __shared__ __bf16 lBlo[16 * LDBF];

    const int tid  = threadIdx.x;
    const int wave = tid >> 5;
    const int lane = tid & 31;
    const int col  = lane & 15;        // N index within tile (B/C/D lane = column)
    const int half = lane >> 4;        // half-wave select
    const int n0   = (blockIdx.x * WAVES + wave) * 16;
    const int mt0  = blockIdx.y * mtiles_per_chunk;

    // Per-wave A tile: split to bf16 hi/lo in LDS (reused across all m-tiles)
    stage_split(A + (size_t)n0 * DDIM, lAhi[wave], lAlo[wave], lane, 32);

    float x2v[8];
#pragma unroll
    for (int v = 0; v < 8; ++v) x2v[v] = x2[n0 + v + 8 * half];

    float rowsum[8];
#pragma unroll
    for (int v = 0; v < 8; ++v) rowsum[v] = 0.0f;

    const __bf16* aRowHi = &lAhi[wave][col * LDBF];
    const __bf16* aRowLo = &lAlo[wave][col * LDBF];
    const __bf16* bRowHi = &lBhi[col * LDBF];
    const __bf16* bRowLo = &lBlo[col * LDBF];

    for (int mt = 0; mt < mtiles_per_chunk; ++mt) {
        const int m0 = (mt0 + mt) * 16;
        if (m0 >= M) break;

        __syncthreads();                               // previous tile fully consumed
        stage_split(B + (size_t)m0 * DDIM, lBhi, lBlo, tid, 32 * WAVES);
        __syncthreads();                               // B tile visible to all waves

        // Speculative prefetch of the next B tile (global_prefetch_b8)
        const int mnext = m0 + 16;
        if (mt + 1 < mtiles_per_chunk && mnext < M) {
            const int pr = tid >> 3, pseg = tid & 7;   // 16 rows x 8 segments
            __builtin_prefetch(B + (size_t)(mnext + pr) * DDIM + pseg * 32, 0, 1);
        }

        v8f c = {0.f, 0.f, 0.f, 0.f, 0.f, 0.f, 0.f, 0.f};
#pragma unroll 2
        for (int k0 = 0; k0 < DDIM; k0 += 32) {
            // A 16x32 bf16 layout: lanes 0-15 -> K = k0..k0+7 (v0-3), k0+16..23 (v4-7);
            //                      lanes 16-31 -> +8 on both chunks.
            v8bf ah0 = *(const v8bf*)&aRowHi[k0 + 8 * half];
            v8bf ah1 = *(const v8bf*)&aRowHi[k0 + 16 + 8 * half];
            v8bf al0 = *(const v8bf*)&aRowLo[k0 + 8 * half];
            v8bf al1 = *(const v8bf*)&aRowLo[k0 + 16 + 8 * half];
            // B 32x16 bf16 layout: lane = column, K = k0+8*half..+7 (v0-3), k0+16+8*half..+7 (v4-7).
            v8bf bh0 = *(const v8bf*)&bRowHi[k0 + 8 * half];
            v8bf bh1 = *(const v8bf*)&bRowHi[k0 + 16 + 8 * half];
            v8bf bl0 = *(const v8bf*)&bRowLo[k0 + 8 * half];
            v8bf bl1 = *(const v8bf*)&bRowLo[k0 + 16 + 8 * half];

            v16bf aH = SHUF16(ah0, ah1);
            v16bf aL = SHUF16(al0, al1);
            v16bf bH = SHUF16(bh0, bh1);
            v16bf bL = SHUF16(bl0, bl1);

            c = __builtin_amdgcn_wmma_f32_16x16x32_bf16(false, aH, false, bH, (short)0, c, false, false);
            c = __builtin_amdgcn_wmma_f32_16x16x32_bf16(false, aH, false, bL, (short)0, c, false, false);
            c = __builtin_amdgcn_wmma_f32_16x16x32_bf16(false, aL, false, bH, (short)0, c, false, false);
        }

        const float y2v = y2[m0 + col];
#pragma unroll
        for (int v = 0; v < 8; ++v) {                  // C/D layout: M = v + 8*half, N = col
            float d2 = x2v[v] + y2v - 2.0f * c[v];
            d2 = fmaxf(d2, 0.0f);
            rowsum[v] += sqrtf(d2 + EPSF);
        }
    }

    // Reduce each row over the 16 lanes of its half-wave, then one atomic/row.
#pragma unroll
    for (int v = 0; v < 8; ++v) {
        float s = rowsum[v];
        s += __shfl_xor(s, 1, 32);
        s += __shfl_xor(s, 2, 32);
        s += __shfl_xor(s, 4, 32);
        s += __shfl_xor(s, 8, 32);
        if (col == 0) atomicAdd(&d_accum[n0 + v + 8 * half], s);
    }
}

// ---------------------------------------------------------------------------
// loss = sum_{i,j} sqrt((d_i - d_j)^2 + eps), d = d_accum / M
// ---------------------------------------------------------------------------
__global__ void pairloss_kernel(const float* __restrict__ d_accum, int N, float invM,
                                float* __restrict__ out) {
    __shared__ float tile[256];
    __shared__ float red[256];
    const int tx = threadIdx.x;
    const int i  = blockIdx.x * 256 + tx;
    const float di = (i < N) ? d_accum[i] * invM : 0.0f;

    float acc = 0.0f;
    for (int j0 = 0; j0 < N; j0 += 256) {
        __syncthreads();
        tile[tx] = (j0 + tx < N) ? d_accum[j0 + tx] * invM : 0.0f;
        __syncthreads();
        if (i < N) {
            for (int jj = 0; jj < 256 && (j0 + jj) < N; ++jj) {
                float diff = di - tile[jj];
                acc += sqrtf(diff * diff + EPSF);
            }
        }
    }
    red[tx] = acc;
    __syncthreads();
    for (int s = 128; s > 0; s >>= 1) {
        if (tx < s) red[tx] += red[tx + s];
        __syncthreads();
    }
    if (tx == 0) atomicAdd(out, red[0]);
}

// ---------------------------------------------------------------------------
extern "C" void kernel_launch(void* const* d_in, const int* in_sizes, int n_in,
                              void* d_out, int out_size, void* d_ws, size_t ws_size,
                              hipStream_t stream) {
    const float* ide = (const float*)d_in[0];
    const float* u   = (const float*)d_in[1];
    float* out = (float*)d_out;

    const int N = in_sizes[0] / DDIM;   // 4096
    const int M = in_sizes[1] / DDIM;   // 16384

    float* x2      = (float*)d_ws;      // [N]
    float* y2      = x2 + N;            // [M]
    float* d_accum = y2 + M;            // [N]

    init_kernel<<<(N + 255) / 256, 256, 0, stream>>>(d_accum, N, out);
    rownorm_kernel<<<N, 256, 0, stream>>>(ide, x2, N);
    rownorm_kernel<<<M, 256, 0, stream>>>(u, y2, M);

    const int mtiles_per_chunk = 16;    // 256 columns of u per block pass
    dim3 grid(N / (16 * WAVES), (M / 16 + mtiles_per_chunk - 1) / mtiles_per_chunk);
    dist_tile_kernel<<<grid, 32 * WAVES, 0, stream>>>(ide, u, x2, y2, d_accum, N, M,
                                                      mtiles_per_chunk);

    pairloss_kernel<<<(N + 255) / 256, 256, 0, stream>>>(d_accum, N, 1.0f / (float)M, out);
}
